// GCNLayer_6889127543192
// MI455X (gfx1250) — matrix operand order repaired and checked by reading, not compiled, not run on hardware
//
#include <hip/hip_runtime.h>
#include <hip/hip_bf16.h>
#include <math.h>

#define N_SCALC 32
#define N_VECC  16
#define N_NODES 8192
#define N_EDGES 65536
#define REMB    64
#define HIDDEN  64
#define WNUMEL  2560
#define OUTD    80
#define NTILES  (N_EDGES / 16)     // 4096 wave tiles
#define NT_W2   160                // 2560 / 16

#define PW2_ELEMS  (NT_W2 * 2 * 32 * 16)      // 163840
#define PW01_ELEMS (2 * 4 * 2 * 32 * 16)      // 8192
#define PEE_ELEMS  (NTILES * 2 * 32 * 16)     // 4194304

typedef __bf16 bf16_t;
typedef __attribute__((ext_vector_type(16))) __bf16 v16bf;
typedef __attribute__((ext_vector_type(8)))  float  v8f;

#define SILU_NORM   1.6790425f
#define PW0         0.14433757f   // sqrt(1/48)
#define PW1         0.21650635f   // sqrt(3/64)
#define INV_SQRT3   0.57735027f
#define INV_SQRT6   0.40824829f
#define WSCALE      0.125f        // 1/sqrt(64)

// 16-bit A-matrix 16x32 layout (ISA 7.12.2): lane holds row m=lane&15,
// lanes 0-15 hold K in [0,8)∪[16,24), lanes 16-31 K in [8,16)∪[24,32).
__device__ __forceinline__ int kmapA(int i, int half) {
  return ((i < 8) ? 0 : 16) + half * 8 + (i & 7);
}
// B-matrix 32x16: lane holds col n=lane&15; same K order as A (lanes 0-15 K in [0,8)∪[16,24), 16-31 K in [8,16)∪[24,32)).
__device__ __forceinline__ int kmapB(int i, int half) {
  return ((i < 8) ? 0 : 16) + half * 8 + (i & 7);
}

// silu(x)*SILU_NORM with fast native rcp (v_rcp_f32) instead of IEEE divide.
__device__ __forceinline__ float silu_n(float x) {
  float den = 1.0f + __expf(-x);
  return x * __builtin_amdgcn_rcpf(den) * SILU_NORM;
}

// ---------------------------------------------------------------------------
// Packing kernels: fp32 -> bf16 in WMMA fragment order (scale folded in).
// Fragment element address = ((frag * 32) + lane) * 16 + i  -> each lane's
// 16 bf16 values are one contiguous 32-byte chunk.
// ---------------------------------------------------------------------------
__global__ __launch_bounds__(256) void pack_w2(const float* __restrict__ W,
                                               bf16_t* __restrict__ P) {
  int gid = blockIdx.x * 256 + threadIdx.x;
  if (gid >= PW2_ELEMS) return;
  int i = gid & 15, l = (gid >> 4) & 31, c = (gid >> 9) & 1, t = gid >> 10;
  int k = c * 32 + kmapB(i, l >> 4);
  int j = t * 16 + (l & 15);
  P[gid] = (bf16_t)(W[(size_t)k * WNUMEL + j] * WSCALE);
}

__global__ __launch_bounds__(256) void pack_w01(const float* __restrict__ W0,
                                                const float* __restrict__ W1,
                                                bf16_t* __restrict__ P) {
  int gid = blockIdx.x * 256 + threadIdx.x;
  if (gid >= PW01_ELEMS) return;
  int i = gid & 15, l = (gid >> 4) & 31, c = (gid >> 9) & 1;
  int nt = (gid >> 10) & 3, layer = gid >> 12;
  const float* W = layer ? W1 : W0;
  int k = c * 32 + kmapB(i, l >> 4);
  int col = nt * 16 + (l & 15);
  P[gid] = (bf16_t)(W[(size_t)k * HIDDEN + col] * WSCALE);
}

__global__ __launch_bounds__(256) void pack_ee(const float* __restrict__ ee,
                                               bf16_t* __restrict__ P) {
  int gid = blockIdx.x * 256 + threadIdx.x;
  if (gid >= PEE_ELEMS) return;
  int i = gid & 15, l = (gid >> 4) & 31, c = (gid >> 9) & 1, tile = gid >> 10;
  int k = c * 32 + kmapA(i, l >> 4);
  int m = l & 15;
  P[gid] = (bf16_t)ee[(size_t)(tile * 16 + m) * REMB + k];
}

// ---------------------------------------------------------------------------
// Kernel: self-connection, also initializes d_out.
// ---------------------------------------------------------------------------
__global__ __launch_bounds__(256) void self_kernel(
    const float* __restrict__ nf, const float* __restrict__ Ws0,
    const float* __restrict__ Ws1, float* __restrict__ out) {
  int idx = blockIdx.x * blockDim.x + threadIdx.x;
  if (idx >= N_NODES * OUTD) return;
  int n = idx / OUTD, j = idx % OUTD;
  const float* row = nf + (size_t)n * OUTD;
  float acc = 0.0f;
  if (j < N_SCALC) {
    #pragma unroll
    for (int u = 0; u < N_SCALC; ++u) acc += row[u] * Ws0[u * N_SCALC + j];
    acc *= 0.17677670f;  // 1/sqrt(32)
  } else {
    int v = (j - N_SCALC) / 3, k = (j - N_SCALC) % 3;
    #pragma unroll
    for (int u = 0; u < N_VECC; ++u) acc += row[N_SCALC + u * 3 + k] * Ws1[u * N_VECC + v];
    acc *= 0.25f;        // 1/sqrt(16)
  }
  out[idx] = acc;
}

// ---------------------------------------------------------------------------
// Main fused kernel: one wave = 16 edges. All WMMA B operands come from
// pre-packed bf16 fragments; inner 160-tile loop has no barriers.
// ---------------------------------------------------------------------------
template <bool PEE>
__global__ __launch_bounds__(32) void edge_kernel(
    const int*   __restrict__ ei,  const float* __restrict__ nf,
    const float* __restrict__ ef,  const float* __restrict__ ee,
    const v16bf* __restrict__ pEE, const v16bf* __restrict__ pW01,
    const v16bf* __restrict__ pW2, float* __restrict__ out) {
  __shared__ float Hbuf[16][68];   // activation transpose buffer
  __shared__ float a0s [16][48];   // [x0*y0 (32) | dot/sqrt3 (16)]
  __shared__ float x0s [16][32];
  __shared__ float x1s [16][48];   // x1[u][k]
  __shared__ float crs [16][48];   // cross(x1,y1)[u][k]
  __shared__ float y0s [16];
  __shared__ float y1s [16][3];
  __shared__ int   dsts[16];

  const int lane  = threadIdx.x & 31;
  const int half  = lane >> 4;
  const int mrow  = lane & 15;
  const int tile  = blockIdx.x;
  const int edge0 = tile * 16;

  // ---- A fragments of the edge-embedding tile (16x64) ----
  v16bf aE0, aE1;
  if (PEE) {
    aE0 = pEE[(tile * 2 + 0) * 32 + lane];
    aE1 = pEE[(tile * 2 + 1) * 32 + lane];
  } else {
    const float* row = ee + (size_t)(edge0 + mrow) * REMB;
    #pragma unroll
    for (int i = 0; i < 16; ++i) {
      aE0[i] = (bf16_t)row[kmapA(i, half)];
      aE1[i] = (bf16_t)row[32 + kmapA(i, half)];
    }
  }

  // ---- GEMM1: H0 = silu(E @ W0/8) * SILU_NORM ----
  #pragma unroll
  for (int nt = 0; nt < 4; ++nt) {
    v16bf b0 = pW01[((0 * 4 + nt) * 2 + 0) * 32 + lane];
    v16bf b1 = pW01[((0 * 4 + nt) * 2 + 1) * 32 + lane];
    v8f d = {};
    d = __builtin_amdgcn_wmma_f32_16x16x32_bf16(false, aE0, false, b0, (short)0, d, false, false);
    d = __builtin_amdgcn_wmma_f32_16x16x32_bf16(false, aE1, false, b1, (short)0, d, false, false);
    #pragma unroll
    for (int r = 0; r < 8; ++r)
      Hbuf[r + 8 * half][nt * 16 + mrow] = silu_n(d[r]);
  }
  __syncthreads();

  // ---- A fragments of H0 ----
  v16bf aM0, aM1;
  #pragma unroll
  for (int i = 0; i < 16; ++i) {
    aM0[i] = (bf16_t)Hbuf[mrow][kmapA(i, half)];
    aM1[i] = (bf16_t)Hbuf[mrow][32 + kmapA(i, half)];
  }
  __syncthreads();

  // ---- GEMM2: H1 = silu(H0 @ W1/8) * SILU_NORM ----
  #pragma unroll
  for (int nt = 0; nt < 4; ++nt) {
    v16bf b0 = pW01[((1 * 4 + nt) * 2 + 0) * 32 + lane];
    v16bf b1 = pW01[((1 * 4 + nt) * 2 + 1) * 32 + lane];
    v8f d = {};
    d = __builtin_amdgcn_wmma_f32_16x16x32_bf16(false, aM0, false, b0, (short)0, d, false, false);
    d = __builtin_amdgcn_wmma_f32_16x16x32_bf16(false, aM1, false, b1, (short)0, d, false, false);
    #pragma unroll
    for (int r = 0; r < 8; ++r)
      Hbuf[r + 8 * half][nt * 16 + mrow] = silu_n(d[r]);
  }
  __syncthreads();

  // ---- A fragments of H1 (live across the whole 160-tile loop) ----
  v16bf aF0, aF1;
  #pragma unroll
  for (int i = 0; i < 16; ++i) {
    aF0[i] = (bf16_t)Hbuf[mrow][kmapA(i, half)];
    aF1[i] = (bf16_t)Hbuf[mrow][32 + kmapA(i, half)];
  }

  // ---- per-edge geometry staging (one lane per edge) ----
  if (lane < 16) {
    int eg = edge0 + lane;
    dsts[lane] = ei[N_EDGES + eg];
    int  s  = ei[eg];
    float y0 = ef[(size_t)eg * 4 + 0];
    float e1 = ef[(size_t)eg * 4 + 1];
    float e2 = ef[(size_t)eg * 4 + 2];
    float e3 = ef[(size_t)eg * 4 + 3];
    y0s[lane] = y0;
    y1s[lane][0] = e1; y1s[lane][1] = e2; y1s[lane][2] = e3;
    const float* row = nf + (size_t)s * OUTD;
    #pragma unroll
    for (int u = 0; u < 32; ++u) {
      float v = row[u];
      x0s[lane][u] = v;
      a0s[lane][u] = v * y0;
    }
    #pragma unroll
    for (int u = 0; u < 16; ++u) {
      float vx = row[32 + u * 3 + 0];
      float vy = row[32 + u * 3 + 1];
      float vz = row[32 + u * 3 + 2];
      x1s[lane][u * 3 + 0] = vx;
      x1s[lane][u * 3 + 1] = vy;
      x1s[lane][u * 3 + 2] = vz;
      a0s[lane][32 + u] = INV_SQRT3 * (vx * e1 + vy * e2 + vz * e3);
      crs[lane][u * 3 + 0] = vy * e3 - vz * e2;
      crs[lane][u * 3 + 1] = vz * e1 - vx * e3;
      crs[lane][u * 3 + 2] = vx * e2 - vy * e1;
    }
  }
  __syncthreads();

  // ---- big GEMM [16x64]@[64x2560] fused with segment contraction ----
  float acc0a[8] = {}, acc0b[8] = {}, acc3[8] = {};
  float acc4[3][8] = {}, acc5[3][8] = {};

  for (int t = 0; t < NT_W2; ++t) {
    v16bf b0 = pW2[(t * 2 + 0) * 32 + lane];
    v16bf b1 = pW2[(t * 2 + 1) * 32 + lane];

    v8f d = {};
    d = __builtin_amdgcn_wmma_f32_16x16x32_bf16(false, aF0, false, b0, (short)0, d, false, false);
    d = __builtin_amdgcn_wmma_f32_16x16x32_bf16(false, aF1, false, b1, (short)0, d, false, false);

    // D[m=r+8*half (edge), n=mrow] == w[e, 16t + n]; contract per segment.
    if (t < 64) {                              // W1 (32x32): u=t/2, w-half=t&1
      int u = t >> 1, hi = t & 1;
      #pragma unroll
      for (int r = 0; r < 8; ++r) {
        float m = a0s[r + 8 * half][u];
        if (hi) acc0b[r] += d[r] * m; else acc0a[r] += d[r] * m;
      }
    } else if (t < 96) {                       // W2 (16x32): dot term
      int lt = t - 64; int u = 32 + (lt >> 1), hi = lt & 1;
      #pragma unroll
      for (int r = 0; r < 8; ++r) {
        float m = a0s[r + 8 * half][u];
        if (hi) acc0b[r] += d[r] * m; else acc0a[r] += d[r] * m;
      }
    } else if (t < 128) {                      // W3 (32x16): x0 ⊗ y1
      int u = t - 96;
      #pragma unroll
      for (int r = 0; r < 8; ++r)
        acc3[r] += d[r] * x0s[r + 8 * half][u];
    } else if (t < 144) {                      // W4 (16x16): x1 * y0
      int u = t - 128;
      #pragma unroll
      for (int r = 0; r < 8; ++r) {
        #pragma unroll
        for (int k = 0; k < 3; ++k)
          acc4[k][r] += d[r] * x1s[r + 8 * half][u * 3 + k];
      }
    } else {                                   // W5 (16x16): cross term
      int u = t - 144;
      #pragma unroll
      for (int r = 0; r < 8; ++r) {
        #pragma unroll
        for (int k = 0; k < 3; ++k)
          acc5[k][r] += d[r] * crs[r + 8 * half][u * 3 + k];
      }
    }
  }

  // ---- finalize + scatter-add (hardware fp32 atomics) ----
  #pragma unroll
  for (int r = 0; r < 8; ++r) {
    int e = r + 8 * half;
    int dn = dsts[e];
    float* ob = out + (size_t)dn * OUTD;
    unsafeAtomicAdd(ob + mrow,      PW0 * acc0a[r]);
    unsafeAtomicAdd(ob + 16 + mrow, PW0 * acc0b[r]);
    float y0 = y0s[e];
    #pragma unroll
    for (int k = 0; k < 3; ++k) {
      float v = PW1 * (INV_SQRT3 * (acc3[r] * y1s[e][k] + acc4[k][r] * y0)
                       + INV_SQRT6 * acc5[k][r]);
      unsafeAtomicAdd(ob + 32 + mrow * 3 + k, v);
    }
  }
}

extern "C" void kernel_launch(void* const* d_in, const int* in_sizes, int n_in,
                              void* d_out, int out_size, void* d_ws, size_t ws_size,
                              hipStream_t stream) {
  (void)in_sizes; (void)n_in; (void)out_size;
  const int*   ei  = (const int*)  d_in[0];
  const float* nf  = (const float*)d_in[1];
  const float* ef  = (const float*)d_in[2];
  const float* ee  = (const float*)d_in[3];
  const float* Wf0 = (const float*)d_in[5];
  const float* Wf1 = (const float*)d_in[6];
  const float* Wf2 = (const float*)d_in[7];
  const float* Ws0 = (const float*)d_in[8];
  const float* Ws1 = (const float*)d_in[9];
  float* out = (float*)d_out;

  bf16_t* pW2  = (bf16_t*)d_ws;           // 163840 elems = 320 KB
  bf16_t* pW01 = pW2 + PW2_ELEMS;         // 8192 elems   =  16 KB
  bf16_t* pEEb = pW01 + PW01_ELEMS;       // 4194304 elems = 8 MB
  size_t need_ee = (size_t)(PW2_ELEMS + PW01_ELEMS + PEE_ELEMS) * sizeof(bf16_t);
  bool use_pee = ws_size >= need_ee;

  pack_w2 <<<(PW2_ELEMS  + 255) / 256, 256, 0, stream>>>(Wf2, pW2);
  pack_w01<<<(PW01_ELEMS + 255) / 256, 256, 0, stream>>>(Wf0, Wf1, pW01);
  if (use_pee)
    pack_ee<<<(PEE_ELEMS + 255) / 256, 256, 0, stream>>>(ee, pEEb);

  self_kernel<<<(N_NODES * OUTD + 255) / 256, 256, 0, stream>>>(nf, Ws0, Ws1, out);

  if (use_pee)
    edge_kernel<true><<<NTILES, 32, 0, stream>>>(ei, nf, ef, ee,
        (const v16bf*)pEEb, (const v16bf*)pW01, (const v16bf*)pW2, out);
  else
    edge_kernel<false><<<NTILES, 32, 0, stream>>>(ei, nf, ef, ee,
        (const v16bf*)pEEb, (const v16bf*)pW01, (const v16bf*)pW2, out);
}
